// RWKV_TimeMixing_84413287235990
// MI455X (gfx1250) — matrix operand order repaired and checked by reading, not compiled, not run on hardware
//
#include <hip/hip_runtime.h>
#include <math.h>

// ---------------------------------------------------------------------------
// RWKV time-mixing for MI455X (gfx1250, wave32).  B=8, T=2048, D=K=1024, H=1.
//
// GEMMs: bf16x3 split precision on v_wmma_f32_16x16x32_bf16, block-tiled
// 64x128 per workgroup (8 waves, 2x4), A/B staged in LDS via
// global_load_async_to_lds_b128 with double buffering (ASYNCcnt).
// Scan: 3-phase chunk-parallel linear recurrence (16 chunks of 128 steps).
// ---------------------------------------------------------------------------

typedef __bf16 bf16;
typedef bf16  v8bf  __attribute__((ext_vector_type(8)));
typedef bf16  v16bf __attribute__((ext_vector_type(16)));
typedef float v8f   __attribute__((ext_vector_type(8)));

#define D_DIM 1024
#define T_DIM 2048
#define B_DIM 8
#define BT_DIM (B_DIM * T_DIM)          // 16384 rows
#define N_ELT ((size_t)BT_DIM * D_DIM)  // 16,777,216
#define W_ELT ((size_t)D_DIM * D_DIM)   // 1,048,576
#define EPS_WKV 1e-9f
#define EPS_GN  1e-5f

// scan chunking
#define N_CH   16
#define L_CH   128                      // T_DIM / N_CH

// LDS tile geometry (elements): padded row stride 40 (80B) for conflict-free
#define A_STRIDE 40
#define A_PART   (64 * A_STRIDE)        // 2560
#define B_STRIDE 40
#define B_PART   (128 * B_STRIDE)       // 5120
#define BUF_ELT  (2 * A_PART + 2 * B_PART)  // 15360 elts = 30720 B per buffer

// ---- bf16 WMMA wrapper ----------------------------------------------------
__device__ __forceinline__ v8f wmma_bf(v16bf a, v16bf b, v8f c) {
  return __builtin_amdgcn_wmma_f32_16x16x32_bf16(
      false, a, false, b, (short)0, c, false, false);
}

__device__ __forceinline__ void split_bf16(float a, bf16& h, bf16& l) {
  h = (bf16)a;
  l = (bf16)(a - (float)h);
}

__device__ __forceinline__ v16bf concat8(v8bf a, v8bf b) {
  return __builtin_shufflevector(a, b, 0, 1, 2, 3, 4, 5, 6, 7,
                                 8, 9, 10, 11, 12, 13, 14, 15);
}

// ---- async global -> LDS (16B per lane), tracked by ASYNCcnt --------------
__device__ __forceinline__ void async_to_lds_b128(const bf16* gaddr,
                                                  const bf16* laddr) {
  unsigned loff = (unsigned)(unsigned long long)laddr;  // LDS = low 32 bits
  asm volatile("global_load_async_to_lds_b128 %0, %1, off"
               :: "v"(loff), "v"(gaddr) : "memory");
}
__device__ __forceinline__ void wait_asynccnt0() {
  asm volatile("s_wait_asynccnt 0x0" ::: "memory");
}

// ---- C/D tile store: M = i + 8*(lane>=16), N = lane&15 --------------------
// mode 0: sigmoid, 1: exp, 2: identity
__device__ __forceinline__ void store_tile(float* __restrict__ out, v8f acc,
                                           int rowbase, int colbase,
                                           int lane, int mode) {
  const int mh = (lane >> 4) << 3;
  const int nc = lane & 15;
#pragma unroll
  for (int i = 0; i < 8; ++i) {
    float v = acc[i];
    if (mode == 0)      v = 1.0f / (1.0f + expf(-v));
    else if (mode == 1) v = expf(v);
    out[(size_t)(rowbase + mh + i) * D_DIM + (colbase + nc)] = v;
  }
}

// ===========================================================================
// [0] Weight split (grid.y selects r/k/v/o)
// ===========================================================================
__global__ __launch_bounds__(256) void prep_w(
    const float* __restrict__ Wr, const float* __restrict__ Wk,
    const float* __restrict__ Wv, const float* __restrict__ Wo,
    bf16* __restrict__ whi, bf16* __restrict__ wlo) {
  const int z = blockIdx.y;
  const float* __restrict__ W =
      (z == 0) ? Wr : (z == 1) ? Wk : (z == 2) ? Wv : Wo;
  const size_t idx = (size_t)blockIdx.x * blockDim.x + threadIdx.x;
  bf16 h, l;
  split_bf16(W[idx], h, l);
  whi[(size_t)z * W_ELT + idx] = h;
  wlo[(size_t)z * W_ELT + idx] = l;
}

// ===========================================================================
// [1] Token-shift mix + bf16 split for xr/xk/xv
// ===========================================================================
__global__ __launch_bounds__(256) void prep_mix(
    const float* __restrict__ x, const float* __restrict__ state,
    const float* __restrict__ mr, const float* __restrict__ mk,
    const float* __restrict__ mv,
    bf16* __restrict__ ahi, bf16* __restrict__ alo) {
  const size_t idx = (size_t)blockIdx.x * blockDim.x + threadIdx.x;
  const int g = (int)(idx >> 10);
  const int d = (int)(idx & (D_DIM - 1));
  const int b = g >> 11;
  const int t = g & (T_DIM - 1);

  const float xv = x[idx];
  const float xp = (t == 0) ? state[((size_t)b * D_DIM + d) * 3 + 2]
                            : x[idx - D_DIM];
  bf16 h, l;
  float a;
  a = fmaf(mr[d], xv - xp, xp);
  split_bf16(a, h, l);
  ahi[idx] = h;               alo[idx] = l;
  a = fmaf(mk[d], xv - xp, xp);
  split_bf16(a, h, l);
  ahi[N_ELT + idx] = h;       alo[N_ELT + idx] = l;
  a = fmaf(mv[d], xv - xp, xp);
  split_bf16(a, h, l);
  ahi[2 * N_ELT + idx] = h;   alo[2 * N_ELT + idx] = l;
}

// ===========================================================================
// [2] r/k/v GEMMs (bf16x3): block tile 64x128, 8 waves (2x4), LDS
//     double-buffered via async copies. grid.y = z.
// ===========================================================================
__global__ __launch_bounds__(256) void rkv_gemm(
    const bf16* __restrict__ ahi, const bf16* __restrict__ alo,
    const bf16* __restrict__ whi, const bf16* __restrict__ wlo,
    float* __restrict__ outR, float* __restrict__ outEK,
    float* __restrict__ outV) {
  __shared__ bf16 smem[2 * BUF_ELT];

  const int tid   = threadIdx.x;
  const int lane  = tid & 31;
  const int widx  = tid >> 5;
  const int waveM = widx >> 2;           // 0..1
  const int waveN = widx & 3;            // 0..3
  const int z     = blockIdx.y;

  const int bm = blockIdx.x >> 3;        // 256 row blocks of 64
  const int bn = blockIdx.x & 7;         // 8 col blocks of 128
  const int r0 = bm << 6;
  const int c0 = bn << 7;

  const bf16* __restrict__ Ah = ahi + (size_t)z * N_ELT;
  const bf16* __restrict__ Al = alo + (size_t)z * N_ELT;
  const bf16* __restrict__ Bh = whi + (size_t)z * W_ELT;
  const bf16* __restrict__ Bl = wlo + (size_t)z * W_ELT;
  float* __restrict__ out = (z == 0) ? outR : ((z == 1) ? outEK : outV);

  const int h = lane >> 4;
  // LDS fragment read offsets (elements)
  const int rA0 = waveM * 32 + (lane & 15);        // frag row, +16 for frag1
  const int cB0 = waveN * 32 + (lane & 15);        // frag col, +16 for frag1

  v8f acc00 = {}, acc01 = {}, acc10 = {}, acc11 = {};

  // cooperative async fill of one buffer for K-chunk kk
  auto fill = [&](int buf, int kk) {
    bf16* base = smem + buf * BUF_ELT;
#pragma unroll
    for (int i = 0; i < 6; ++i) {
      const int u = tid + (i << 8);
      const bf16* g;
      int loff;
      if (u < 512) {  // A: part(2) x row(64) x seg(4 x 16B)
        const int p = u >> 8, row = (u >> 2) & 63, seg = u & 3;
        g = (p ? Al : Ah) + (size_t)(r0 + row) * D_DIM + kk + seg * 8;
        loff = p * A_PART + row * A_STRIDE + seg * 8;
      } else {        // B: part(2) x col(128) x seg(4 x 16B)
        const int ub = u - 512;
        const int p = ub >> 9, col = (ub >> 2) & 127, seg = ub & 3;
        g = (p ? Bl : Bh) + (size_t)(c0 + col) * D_DIM + kk + seg * 8;
        loff = 2 * A_PART + p * B_PART + col * B_STRIDE + seg * 8;
      }
      async_to_lds_b128(g, base + loff);
    }
  };

  fill(0, 0);

  for (int c = 0; c < D_DIM / 32; ++c) {
    const int cur = c & 1;
    wait_asynccnt0();
    __syncthreads();
    if (c + 1 < D_DIM / 32) fill(cur ^ 1, (c + 1) * 32);

    const bf16* buf = smem + cur * BUF_ELT;
    const bf16* ap0 = buf + rA0 * A_STRIDE + 16 * h;           // hi part
    const bf16* ap1 = buf + A_PART + rA0 * A_STRIDE + 16 * h;  // lo part
    const bf16* bp0 = buf + 2 * A_PART + cB0 * B_STRIDE + 16 * h;
    const bf16* bp1 = buf + 2 * A_PART + B_PART + cB0 * B_STRIDE + 16 * h;

    v16bf ah0 = concat8(*(const v8bf*)(ap0),
                        *(const v8bf*)(ap0 + 8));
    v16bf al0 = concat8(*(const v8bf*)(ap1),
                        *(const v8bf*)(ap1 + 8));
    v16bf ah1 = concat8(*(const v8bf*)(ap0 + 16 * A_STRIDE),
                        *(const v8bf*)(ap0 + 16 * A_STRIDE + 8));
    v16bf al1 = concat8(*(const v8bf*)(ap1 + 16 * A_STRIDE),
                        *(const v8bf*)(ap1 + 16 * A_STRIDE + 8));
    v16bf bh0 = *(const v16bf*)(bp0);
    v16bf bl0 = *(const v16bf*)(bp1);
    v16bf bh1 = *(const v16bf*)(bp0 + 16 * B_STRIDE);
    v16bf bl1 = *(const v16bf*)(bp1 + 16 * B_STRIDE);

    acc00 = wmma_bf(ah0, bh0, acc00);
    acc00 = wmma_bf(ah0, bl0, acc00);
    acc00 = wmma_bf(al0, bh0, acc00);
    acc01 = wmma_bf(ah0, bh1, acc01);
    acc01 = wmma_bf(ah0, bl1, acc01);
    acc01 = wmma_bf(al0, bh1, acc01);
    acc10 = wmma_bf(ah1, bh0, acc10);
    acc10 = wmma_bf(ah1, bl0, acc10);
    acc10 = wmma_bf(al1, bh0, acc10);
    acc11 = wmma_bf(ah1, bh1, acc11);
    acc11 = wmma_bf(ah1, bl1, acc11);
    acc11 = wmma_bf(al1, bh1, acc11);
    __syncthreads();
  }

  const int wr0 = r0 + waveM * 32;
  const int wc0 = c0 + waveN * 32;
  store_tile(out, acc00, wr0,      wc0,      lane, z);
  store_tile(out, acc01, wr0,      wc0 + 16, lane, z);
  store_tile(out, acc10, wr0 + 16, wc0,      lane, z);
  store_tile(out, acc11, wr0 + 16, wc0 + 16, lane, z);
}

// ===========================================================================
// [3a] Scan phase A: per-chunk partial sums (affine recurrence summaries).
//      grid = (chunk=16, cgroup=4, b=8), 256 thr -> channel.
// ===========================================================================
__global__ __launch_bounds__(256) void scan_phaseA(
    const float* __restrict__ EK, const float* __restrict__ V,
    const float* __restrict__ decay,
    float* __restrict__ Snum, float* __restrict__ Sden) {
  const int j = blockIdx.x;                       // chunk
  const int c = blockIdx.y * 256 + threadIdx.x;   // channel
  const int b = blockIdx.z;
  const float w = expf(-expf(decay[c]));

  float sn = 0.f, sd = 0.f;
  size_t idx = ((size_t)b * T_DIM + j * L_CH) * D_DIM + c;
#pragma unroll 4
  for (int t = 0; t < L_CH; ++t, idx += D_DIM) {
    const float ekt = EK[idx];
    sn = fmaf(sn, w, ekt * V[idx]);
    sd = fmaf(sd, w, ekt);
  }
  const size_t o = ((size_t)b * D_DIM + c) * N_CH + j;
  Snum[o] = sn;
  Sden[o] = sd;
}

// ===========================================================================
// [3b] Scan phase B: sequential combine of chunk summaries -> per-chunk
//      incoming states. One thread per (b, channel).
// ===========================================================================
__global__ __launch_bounds__(256) void scan_phaseB(
    const float* __restrict__ Snum, const float* __restrict__ Sden,
    const float* __restrict__ decay, const float* __restrict__ state,
    float* __restrict__ Inum, float* __restrict__ Iden) {
  const int tid = blockIdx.x * blockDim.x + threadIdx.x;  // 0..8191
  const int b = tid >> 10;
  const int c = tid & (D_DIM - 1);
  const float w  = expf(-expf(decay[c]));
  const float wL = powf(w, (float)L_CH);

  const float* sb = state + ((size_t)b * D_DIM + c) * 3;
  float n = sb[0], d = sb[1];
  const size_t o = ((size_t)b * D_DIM + c) * N_CH;
#pragma unroll
  for (int j = 0; j < N_CH; ++j) {
    Inum[o + j] = n;
    Iden[o + j] = d;
    n = fmaf(wL, n, Snum[o + j]);
    d = fmaf(wL, d, Sden[o + j]);
  }
}

// ===========================================================================
// [3c] Scan phase C: replay each chunk from incoming state; emit rwkv and
//      deterministic per-block GN partials (LDS tree reduction).
// ===========================================================================
__global__ __launch_bounds__(256) void scan_phaseC(
    const float* __restrict__ R, const float* __restrict__ EK,
    const float* __restrict__ V,
    const float* __restrict__ decay, const float* __restrict__ first,
    const float* __restrict__ Inum, const float* __restrict__ Iden,
    float* __restrict__ rwkv, float* __restrict__ partials) {
  __shared__ float rs[256], rq[256];
  const int j = blockIdx.x;
  const int c = blockIdx.y * 256 + threadIdx.x;
  const int b = blockIdx.z;
  const float w  = expf(-expf(decay[c]));
  const float eu = expf(first[c]);

  const size_t o = ((size_t)b * D_DIM + c) * N_CH + j;
  float num = Inum[o], den = Iden[o];
  float lsum = 0.f, lsq = 0.f;

  size_t idx = ((size_t)b * T_DIM + j * L_CH) * D_DIM + c;
#pragma unroll 4
  for (int t = 0; t < L_CH; ++t, idx += D_DIM) {
    const float ekt = EK[idx];
    const float vt  = V[idx];
    const float bonus = eu * ekt;
    const float wkv = (num + bonus * vt) / (den + bonus + EPS_WKV);
    const float ov = R[idx] * wkv;
    rwkv[idx] = ov;
    lsum += ov;
    lsq = fmaf(ov, ov, lsq);
    num = fmaf(num, w, ekt * vt);
    den = fmaf(den, w, ekt);
  }

  rs[threadIdx.x] = lsum;
  rq[threadIdx.x] = lsq;
  __syncthreads();
#pragma unroll
  for (int s = 128; s > 0; s >>= 1) {
    if (threadIdx.x < (unsigned)s) {
      rs[threadIdx.x] += rs[threadIdx.x + s];
      rq[threadIdx.x] += rq[threadIdx.x + s];
    }
    __syncthreads();
  }
  if (threadIdx.x == 0) {
    const int pb = (b * 4 + blockIdx.y) * N_CH + j;  // 64 partials per batch
    partials[pb * 2 + 0] = rs[0];
    partials[pb * 2 + 1] = rq[0];
  }
}

// ===========================================================================
// [4] GroupNorm finalize: deterministic sum of 64 partials per batch.
// ===========================================================================
__global__ void gn_finalize(const float* __restrict__ partials,
                            float* __restrict__ muinv) {
  const int b = threadIdx.x;
  if (b < B_DIM) {
    float s = 0.f, q = 0.f;
    for (int j = 0; j < 64; ++j) {
      s += partials[(b * 64 + j) * 2 + 0];
      q += partials[(b * 64 + j) * 2 + 1];
    }
    const float inv_n = 1.0f / ((float)D_DIM * (float)T_DIM);
    const float mu  = s * inv_n;
    const float var = q * inv_n - mu * mu;
    muinv[b * 2 + 0] = mu;
    muinv[b * 2 + 1] = rsqrtf(var + EPS_GN);
  }
}

// ===========================================================================
// [5] Output GEMM: A[g,d] = rwkv*s[d] + t[d], s = inv*gamma, t = beta - mu*s,
//     split to bf16 on the fly (conversion co-executes with WMMA).
// ===========================================================================
__global__ __launch_bounds__(256) void out_gemm(
    const float* __restrict__ rwkv,
    const bf16* __restrict__ whi, const bf16* __restrict__ wlo,
    const float* __restrict__ gamma, const float* __restrict__ beta,
    const float* __restrict__ muinv, float* __restrict__ out) {
  const int lane = threadIdx.x & 31;
  const int wid  = blockIdx.x * (blockDim.x >> 5) + (threadIdx.x >> 5);
  const int tileN = wid & 31;
  const int tileM = wid >> 5;

  const int r0 = tileM << 5;
  const int c0 = tileN << 5;
  const int bidx = r0 >> 11;            // whole 32-row tile is in one batch
  const float mu  = muinv[bidx * 2 + 0];
  const float inv = muinv[bidx * 2 + 1];

  const int h = lane >> 4;
  const size_t rowA0 = (size_t)(r0 + (lane & 15)) * D_DIM;
  const size_t rowA1 = rowA0 + (size_t)16 * D_DIM;
  const size_t rowB0 = (size_t)(c0 + (lane & 15)) * D_DIM + 3 * W_ELT;
  const size_t rowB1 = rowB0 + (size_t)16 * D_DIM;

  v8f acc00 = {}, acc01 = {}, acc10 = {}, acc11 = {};

#pragma unroll 2
  for (int kk = 0; kk < D_DIM; kk += 32) {
    const int ka = kk + 16 * h;
    const int kb = kk + 16 * h;

    float s[16], t[16], a0[16], a1[16];
#pragma unroll
    for (int r = 0; r < 2; ++r) {
      const int kbase = ka + r * 8;
#pragma unroll
      for (int e = 0; e < 8; ++e) {
        const float gg = gamma[kbase + e] * inv;
        s[r * 8 + e] = gg;
        t[r * 8 + e] = fmaf(-mu, gg, beta[kbase + e]);
        a0[r * 8 + e] = rwkv[rowA0 + kbase + e];
        a1[r * 8 + e] = rwkv[rowA1 + kbase + e];
      }
    }
    v16bf ah0, al0, ah1, al1;
#pragma unroll
    for (int e = 0; e < 16; ++e) {
      bf16 hh, ll;
      split_bf16(fmaf(a0[e], s[e], t[e]), hh, ll);
      ah0[e] = hh; al0[e] = ll;
      split_bf16(fmaf(a1[e], s[e], t[e]), hh, ll);
      ah1[e] = hh; al1[e] = ll;
    }
    v16bf bh0 = *(const v16bf*)(whi + rowB0 + kb);
    v16bf bl0 = *(const v16bf*)(wlo + rowB0 + kb);
    v16bf bh1 = *(const v16bf*)(whi + rowB1 + kb);
    v16bf bl1 = *(const v16bf*)(wlo + rowB1 + kb);

    acc00 = wmma_bf(ah0, bh0, acc00);
    acc00 = wmma_bf(ah0, bl0, acc00);
    acc00 = wmma_bf(al0, bh0, acc00);
    acc01 = wmma_bf(ah0, bh1, acc01);
    acc01 = wmma_bf(ah0, bl1, acc01);
    acc01 = wmma_bf(al0, bh1, acc01);
    acc10 = wmma_bf(ah1, bh0, acc10);
    acc10 = wmma_bf(ah1, bl0, acc10);
    acc10 = wmma_bf(al1, bh0, acc10);
    acc11 = wmma_bf(ah1, bh1, acc11);
    acc11 = wmma_bf(ah1, bl1, acc11);
    acc11 = wmma_bf(al1, bh1, acc11);
  }

  store_tile(out, acc00, r0,      c0,      lane, 2);
  store_tile(out, acc01, r0,      c0 + 16, lane, 2);
  store_tile(out, acc10, r0 + 16, c0,      lane, 2);
  store_tile(out, acc11, r0 + 16, c0 + 16, lane, 2);
}

// ===========================================================================
extern "C" void kernel_launch(void* const* d_in, const int* in_sizes, int n_in,
                              void* d_out, int out_size, void* d_ws,
                              size_t ws_size, hipStream_t stream) {
  (void)in_sizes; (void)n_in; (void)out_size; (void)ws_size;

  const float* x      = (const float*)d_in[0];
  const float* state  = (const float*)d_in[1];
  const float* W_r    = (const float*)d_in[2];
  const float* W_k    = (const float*)d_in[3];
  const float* W_v    = (const float*)d_in[4];
  const float* W_o    = (const float*)d_in[5];
  const float* mix_r  = (const float*)d_in[6];
  const float* mix_k  = (const float*)d_in[7];
  const float* mix_v  = (const float*)d_in[8];
  const float* decay  = (const float*)d_in[9];
  const float* first  = (const float*)d_in[10];
  const float* gammap = (const float*)d_in[11];
  const float* betap  = (const float*)d_in[12];
  float* out = (float*)d_out;

  // Workspace layout
  char* p = (char*)d_ws;
  float* R     = (float*)p;            p += N_ELT * 4;
  float* EK    = (float*)p;            p += N_ELT * 4;
  float* V     = (float*)p;            p += N_ELT * 4;
  float* RW    = (float*)p;            p += N_ELT * 4;
  const size_t CH_ELT = (size_t)B_DIM * D_DIM * N_CH;   // 131072
  float* Snum  = (float*)p;            p += CH_ELT * 4;
  float* Sden  = (float*)p;            p += CH_ELT * 4;
  float* Inum  = (float*)p;            p += CH_ELT * 4;
  float* Iden  = (float*)p;            p += CH_ELT * 4;
  float* parts = (float*)p;            p += B_DIM * 64 * 2 * 4;
  float* muinv = (float*)p;            p += 64;
  bf16* ahi    = (bf16*)p;             p += 3 * N_ELT * 2;
  bf16* alo    = (bf16*)p;             p += 3 * N_ELT * 2;
  bf16* whi    = (bf16*)p;             p += 4 * W_ELT * 2;
  bf16* wlo    = (bf16*)p;

  // [0] weight hi/lo split
  prep_w<<<dim3((unsigned)(W_ELT / 256), 4), 256, 0, stream>>>(
      W_r, W_k, W_v, W_o, whi, wlo);

  // [1] token-shift mix + split
  prep_mix<<<dim3((unsigned)(N_ELT / 256)), 256, 0, stream>>>(
      x, state, mix_r, mix_k, mix_v, ahi, alo);

  // [2] r/k/v GEMMs: 2048 blocks x 3, block tile 64x128
  rkv_gemm<<<dim3(2048, 3), 256, 0, stream>>>(ahi, alo, whi, wlo, R, EK, V);

  // [3] chunk-parallel WKV scan
  scan_phaseA<<<dim3(N_CH, 4, B_DIM), 256, 0, stream>>>(EK, V, decay,
                                                        Snum, Sden);
  scan_phaseB<<<dim3((B_DIM * D_DIM) / 256), 256, 0, stream>>>(
      Snum, Sden, decay, state, Inum, Iden);
  scan_phaseC<<<dim3(N_CH, 4, B_DIM), 256, 0, stream>>>(
      R, EK, V, decay, first, Inum, Iden, RW, parts);

  // [4] finalize GroupNorm stats (deterministic)
  gn_finalize<<<1, 32, 0, stream>>>(parts, muinv);

  // [5] output projection with fused normalization
  out_gemm<<<dim3(2048), 256, 0, stream>>>(RW, whi, wlo, gammap, betap,
                                           muinv, out);
}
